// GraphSpatiotemporalEncoder_38190849196160
// MI455X (gfx1250) — matrix operand order repaired and checked by reading, not compiled, not run on hardware
//
#include <hip/hip_runtime.h>
#include <hip/hip_bf16.h>
#include <math.h>

typedef __attribute__((ext_vector_type(16))) _Float16 v16h;
typedef __attribute__((ext_vector_type(8)))  float    v8f;

// ---- problem constants (match reference) ----
constexpr int cT   = 6;
constexpr int cN   = 3200;
constexpr int cE   = 32000;
constexpr int cB   = 8;
constexpr int cIN  = 128;
constexpr int cG   = 128;
constexpr int cTOK = 512;
constexpr int cGH  = 4;      // GAT heads
constexpr int cNH  = 8;      // transformer heads
constexpr int cHD  = 64;     // TOK / NH
constexpr int cLG  = 3;      // GAT layers
constexpr int cLT  = 2;      // transformer layers
constexpr float NEG_SLOPE = 0.2f;

// ---- float ordering encode for atomic max ----
__device__ __forceinline__ unsigned f2ord(float f) {
  unsigned u = __float_as_uint(f);
  return (u & 0x80000000u) ? ~u : (u | 0x80000000u);
}
__device__ __forceinline__ float ord2f(unsigned u) {
  unsigned v = (u & 0x80000000u) ? (u & 0x7fffffffu) : ~u;
  return __uint_as_float(v);
}
// f2ord(-inf) == 0x007FFFFF
#define ORD_NEG_INF 0x007FFFFFu

union AFrag { v16h v; _Float16 e[16]; };
union CFrag { v8f  v; float    e[8];  };

// LDS byte-offset of a generic pointer known to live in LDS
#define LDS_OFF(p) ((unsigned)(size_t)(__attribute__((address_space(3))) const void*)(p))

#define USE_ASYNC_LDS 1

// per-lane async copy of 16 bytes global -> LDS (ASYNCcnt-tracked, CDNA5)
__device__ __forceinline__ void async_copy_b128(const float* gsrc, unsigned lds_byte_off) {
#if USE_ASYNC_LDS
  asm volatile("global_load_async_to_lds_b128 %0, %1, off"
               :: "v"(lds_byte_off), "v"((unsigned long long)(size_t)gsrc)
               : "memory");
#else
  // synchronous fallback
  float4 d = *(const float4*)gsrc;
  *(float4*)((char*)__builtin_amdgcn_groupstaticsize() /*unused*/, (void)0, (float4*)nullptr) = d;
#endif
}

__device__ __forceinline__ void wait_async_all() {
#if USE_ASYNC_LDS
  asm volatile("s_wait_asynccnt 0" ::: "memory");
#endif
}

// ============================================================
// WMMA GEMM: C[M,N] = act(A[M,K] @ W[K,N] + bias[N])
// 256-thread block (8 waves) computes a 128x64 output tile.
// Wave w owns rows [w*16, w*16+16) x 64 cols (4 accumulators).
// K stepped by 32; f32 tiles staged in LDS via async-to-LDS,
// converted to f16 at fragment-read time, f32 accumulate.
// Requires K%32==0, N%64==0; ragged M handled by clamp+guard.
// ============================================================
__global__ void __launch_bounds__(256)
gemm_wmma_kernel(const float* __restrict__ A,
                 const float* __restrict__ W,
                 const float* __restrict__ bias,
                 float* __restrict__ C,
                 int M, int K, int Nc, int relu) {
  __shared__ float Atile[128 * 32];   // 16 KB
  __shared__ float Wtile[32 * 64];    //  8 KB

  const int tid  = threadIdx.x;
  const int wave = tid >> 5;
  const int lane = tid & 31;
  const int half = lane >> 4;
  const int l16  = lane & 15;
  const int row0 = blockIdx.y * 128;
  const int col0 = blockIdx.x * 64;

  CFrag acc[4];
  #pragma unroll
  for (int nt = 0; nt < 4; ++nt) acc[nt].v = (v8f){};

  for (int k0 = 0; k0 < K; k0 += 32) {
    // ---- stage A tile: 128 rows x 32 K floats = 1024 b128 chunks ----
    #pragma unroll
    for (int i = 0; i < 4; ++i) {
      int ci  = tid * 4 + i;          // 0..1023
      int r   = ci >> 3;              // row in tile
      int seg = ci & 7;               // 4-float segment
      int gr  = row0 + r; gr = (gr < M) ? gr : (M - 1);   // clamp: keep addr valid
      async_copy_b128(A + (size_t)gr * K + k0 + seg * 4,
                      LDS_OFF(&Atile[r * 32 + seg * 4]));
    }
    // ---- stage W tile: 32 K rows x 64 cols = 512 b128 chunks ----
    #pragma unroll
    for (int i = 0; i < 2; ++i) {
      int ci  = tid * 2 + i;          // 0..511
      int r   = ci >> 4;
      int seg = ci & 15;
      async_copy_b128(W + (size_t)(k0 + r) * Nc + col0 + seg * 4,
                      LDS_OFF(&Wtile[r * 64 + seg * 4]));
    }
    wait_async_all();                 // own wave's ASYNCcnt -> 0
    __syncthreads();                  // all waves' deposits visible

    // ---- A fragment (shared across 4 column tiles) ----
    AFrag a;
    const int ar = wave * 16 + l16;
    #pragma unroll
    for (int j = 0; j < 16; ++j) {
      int kk = (j < 8 ? j : j + 8) + half * 8;   // ISA 7.12.2 16-bit A 16x32 layout
      a.e[j] = (_Float16)Atile[ar * 32 + kk];
    }
    // ---- 4 back-to-back WMMAs ----
    #pragma unroll
    for (int nt = 0; nt < 4; ++nt) {
      AFrag b;
      #pragma unroll
      for (int j = 0; j < 16; ++j)               // B 32x16: same K order as A (K 8h+j, 16+8h+j-8)
        b.e[j] = (_Float16)Wtile[((j < 8 ? j : j + 8) + half * 8) * 64 + nt * 16 + l16];
      acc[nt].v = __builtin_amdgcn_wmma_f32_16x16x32_f16(
          false, a.v, false, b.v, (short)0, acc[nt].v, false, false);
    }
    __syncthreads();                  // tiles reusable next K step
  }

  // ---- writeback (guarded for ragged M) ----
  #pragma unroll
  for (int nt = 0; nt < 4; ++nt) {
    int col = col0 + nt * 16 + l16;
    float bv = bias ? bias[col] : 0.0f;
    #pragma unroll
    for (int r = 0; r < 8; ++r) {
      int m = row0 + wave * 16 + r + half * 8;   // C/D: VGPR r -> M=r | 8+r
      if (m < M) {
        float v = acc[nt].e[r] + bv;
        if (relu) v = fmaxf(v, 0.0f);
        C[(size_t)m * Nc + col] = v;
      }
    }
  }
}

// ============================================================
// Fills
// ============================================================
__global__ void fill_f32_kernel(float* p, float v, int n) {
  int i = blockIdx.x * blockDim.x + threadIdx.x;
  if (i < n) p[i] = v;
}
__global__ void fill_u32_kernel(unsigned* p, unsigned v, int n) {
  int i = blockIdx.x * blockDim.x + threadIdx.x;
  if (i < n) p[i] = v;
}

// ============================================================
// GAT edge kernels (batched over T)
// ============================================================
__global__ void edge_logit_kernel(const int* __restrict__ ei,
                                  const float* __restrict__ ea,
                                  const float* __restrict__ we,    // [2, GH*G]  (layer slice)
                                  const float* __restrict__ att,   // [GH, G]    (layer slice)
                                  const float* __restrict__ xl,
                                  const float* __restrict__ xr,
                                  float* __restrict__ logit,
                                  unsigned* __restrict__ lmax) {
  int idx = blockIdx.x * blockDim.x + threadIdx.x;
  if (idx >= cT * cE * cGH) return;
  int h = idx % cGH;
  int e = (idx / cGH) % cE;
  int t = idx / (cGH * cE);
  int src = ei[(size_t)t * 2 * cE + e];
  int dst = ei[(size_t)t * 2 * cE + cE + e];
  float ea0 = ea[((size_t)t * cE + e) * 2 + 0];
  float ea1 = ea[((size_t)t * cE + e) * 2 + 1];
  const float* xls  = xl + ((size_t)t * cN + src) * (cGH * cG) + h * cG;
  const float* xrd  = xr + ((size_t)t * cN + dst) * (cGH * cG) + h * cG;
  const float* we0  = we + h * cG;
  const float* we1  = we + cGH * cG + h * cG;
  const float* attp = att + h * cG;
  float s = 0.f;
  #pragma unroll 4
  for (int c = 0; c < cG; ++c) {
    float m = xls[c] + xrd[c] + ea0 * we0[c] + ea1 * we1[c];
    m = (m > 0.f) ? m : NEG_SLOPE * m;
    s += m * attp[c];
  }
  logit[idx] = s;
  atomicMax(&lmax[((size_t)t * cN + dst) * cGH + h], f2ord(s));
}

__global__ void edge_p_kernel(const int* __restrict__ ei,
                              float* __restrict__ logit,
                              const unsigned* __restrict__ lmax,
                              float* __restrict__ denom) {
  int idx = blockIdx.x * blockDim.x + threadIdx.x;
  if (idx >= cT * cE * cGH) return;
  int h = idx % cGH;
  int e = (idx / cGH) % cE;
  int t = idx / (cGH * cE);
  int dst = ei[(size_t)t * 2 * cE + cE + e];
  float lm = ord2f(lmax[((size_t)t * cN + dst) * cGH + h]);
  float p = __expf(logit[idx] - lm);
  logit[idx] = p;
  atomicAdd(&denom[((size_t)t * cN + dst) * cGH + h], p);
}

__global__ void edge_aggr_kernel(const int* __restrict__ ei,
                                 const float* __restrict__ p,
                                 const float* __restrict__ denom,
                                 const float* __restrict__ xl,
                                 float* __restrict__ acc) {
  int te = blockIdx.x;               // 0 .. T*E-1
  int t = te / cE, e = te % cE;
  int c = threadIdx.x;               // 0..127
  int src = ei[(size_t)t * 2 * cE + e];
  int dst = ei[(size_t)t * 2 * cE + cE + e];
  const float* xls = xl  + ((size_t)t * cN + src) * (cGH * cG);
  float*       ap  = acc + ((size_t)t * cN + dst) * (cGH * cG);
  #pragma unroll
  for (int h = 0; h < cGH; ++h) {
    float alpha = p[((size_t)te) * cGH + h] /
                  (denom[((size_t)t * cN + dst) * cGH + h] + 1e-16f);
    atomicAdd(&ap[h * cG + c], xls[h * cG + c] * alpha);
  }
}

__global__ void node_update_kernel(const float* __restrict__ acc,
                                   const float* __restrict__ bias,
                                   const float* __restrict__ lng,
                                   const float* __restrict__ lnb,
                                   float* __restrict__ h) {
  int node = blockIdx.x;             // 0 .. T*N-1
  int c = threadIdx.x;               // 0..127
  float o = 0.f;
  #pragma unroll
  for (int hh = 0; hh < cGH; ++hh) o += acc[(size_t)node * (cGH * cG) + hh * cG + c];
  o = o * (1.0f / cGH) + bias[c];
  o = fmaxf(o, 0.f);
  float v = o + h[(size_t)node * cG + c];
  __shared__ float s[cG];
  s[c] = v; __syncthreads();
  for (int st = cG / 2; st > 0; st >>= 1) { if (c < st) s[c] += s[c + st]; __syncthreads(); }
  float mu = s[0] * (1.f / cG); __syncthreads();
  float d = v - mu;
  s[c] = d * d; __syncthreads();
  for (int st = cG / 2; st > 0; st >>= 1) { if (c < st) s[c] += s[c + st]; __syncthreads(); }
  float var = s[0] * (1.f / cG);
  h[(size_t)node * cG + c] = d * rsqrtf(var + 1e-5f) * lng[c] + lnb[c];
}

// ============================================================
// Pooling
// ============================================================
__global__ void count_kernel(const int* __restrict__ batch, int* __restrict__ counts) {
  int i = blockIdx.x * blockDim.x + threadIdx.x;
  if (i < cN) atomicAdd(&counts[batch[i]], 1);
}

__global__ void pool_kernel(const float* __restrict__ h, const int* __restrict__ batch,
                            float* __restrict__ meansum, unsigned* __restrict__ maxenc) {
  int node = blockIdx.x;             // t*N+n
  int t = node / cN, n = node % cN;
  int c = threadIdx.x;
  int b = batch[n];
  float v = h[(size_t)node * cG + c];
  atomicAdd(&meansum[((size_t)(t * cB + b)) * cG + c], v);
  atomicMax(&maxenc[((size_t)(t * cB + b)) * cG + c], f2ord(v));
}

__global__ void pool_fin_kernel(const float* __restrict__ meansum,
                                const unsigned* __restrict__ maxenc,
                                const int* __restrict__ counts,
                                float* __restrict__ pooled) {
  int i = blockIdx.x * blockDim.x + threadIdx.x;
  if (i >= cT * cB * 2 * cG) return;
  int c = i % (2 * cG);
  int tb = i / (2 * cG);
  int b = tb % cB;
  float cnt = fmaxf((float)counts[b], 1.f);
  pooled[i] = (c < cG) ? meansum[(size_t)tb * cG + c] / cnt
                       : ord2f(maxenc[(size_t)tb * cG + (c - cG)]);
}

// tokens row layout is (t*B+b); app/hbuf row layout is (b*T+t)
__global__ void build_app_kernel(const float* __restrict__ tokens,
                                 float* __restrict__ app, float* __restrict__ hbuf) {
  int i = blockIdx.x * blockDim.x + threadIdx.x;
  if (i >= cB * cT * cTOK) return;
  int c = i % cTOK;
  int r = i / cTOK;                  // t*B+b
  int t = r / cB, b = r % cB;
  float v = tokens[i];
  size_t o = ((size_t)(b * cT + t)) * cTOK + c;
  app[o] = v; hbuf[o] = v;
}

// ============================================================
// Transformer
// ============================================================
__global__ void attn_kernel(const float* __restrict__ qkv, float* __restrict__ attnout) {
  int idx = blockIdx.x * blockDim.x + threadIdx.x;
  if (idx >= cB * cNH * cT) return;
  int q  = idx % cT;
  int hh = (idx / cT) % cNH;
  int b  = idx / (cT * cNH);
  const float* qp = qkv + ((size_t)(b * cT + q)) * (3 * cTOK) + hh * cHD;
  float sc[cT];
  float mx = -1e30f;
  for (int kt = 0; kt < cT; ++kt) {
    const float* kp = qkv + ((size_t)(b * cT + kt)) * (3 * cTOK) + cTOK + hh * cHD;
    float s = 0.f;
    #pragma unroll 8
    for (int d = 0; d < cHD; ++d) s += qp[d] * kp[d];
    s *= 0.125f;                     // 1/sqrt(64)
    sc[kt] = s; mx = fmaxf(mx, s);
  }
  float den = 0.f;
  for (int kt = 0; kt < cT; ++kt) { sc[kt] = __expf(sc[kt] - mx); den += sc[kt]; }
  float inv = 1.f / den;
  for (int d = 0; d < cHD; ++d) {
    float o = 0.f;
    for (int kt = 0; kt < cT; ++kt)
      o += sc[kt] * qkv[((size_t)(b * cT + kt)) * (3 * cTOK) + 2 * cTOK + hh * cHD + d];
    attnout[((size_t)(b * cT + q)) * cTOK + hh * cHD + d] = o * inv;
  }
}

__global__ void add_ln512_kernel(float* __restrict__ hbuf, const float* __restrict__ add,
                                 const float* __restrict__ g, const float* __restrict__ bb) {
  int row = blockIdx.x;
  int c = threadIdx.x;               // 0..511
  float v = hbuf[(size_t)row * cTOK + c] + add[(size_t)row * cTOK + c];
  __shared__ float s[cTOK];
  s[c] = v; __syncthreads();
  for (int st = cTOK / 2; st > 0; st >>= 1) { if (c < st) s[c] += s[c + st]; __syncthreads(); }
  float mu = s[0] * (1.f / cTOK); __syncthreads();
  float d = v - mu;
  s[c] = d * d; __syncthreads();
  for (int st = cTOK / 2; st > 0; st >>= 1) { if (c < st) s[c] += s[c + st]; __syncthreads(); }
  float var = s[0] * (1.f / cTOK);
  hbuf[(size_t)row * cTOK + c] = d * rsqrtf(var + 1e-5f) * g[c] + bb[c];
}

__global__ void out_kernel(const float* __restrict__ hbuf, const float* __restrict__ app,
                           float* __restrict__ out) {
  int i = blockIdx.x * blockDim.x + threadIdx.x;
  const int t1 = cB * cT * cTOK;            // 24576
  const int t2 = cB * (cT - 1) * 256;       // 10240
  if (i < t1) {
    out[i] = hbuf[i];
  } else if (i < t1 + t2) {
    int j = i - t1;
    int c = j % 256;
    int dt = (j / 256) % (cT - 1);
    int b = j / (256 * (cT - 1));
    out[i] = app[((size_t)(b * cT + dt + 1)) * cTOK + c]
           - app[((size_t)(b * cT + dt)) * cTOK + c];
  }
}

// ============================================================
// Host-side orchestration
// ============================================================
static inline dim3 gemm_grid(int M, int Nc) { return dim3(Nc / 64, (M + 127) / 128); }

extern "C" void kernel_launch(void* const* d_in, const int* in_sizes, int n_in,
                              void* d_out, int out_size, void* d_ws, size_t ws_size,
                              hipStream_t stream) {
  (void)in_sizes; (void)n_in; (void)out_size; (void)ws_size;
  const float* x         = (const float*)d_in[0];
  const int*   edge_idx  = (const int*)d_in[1];
  const float* edge_attr = (const float*)d_in[2];
  const int*   batch     = (const int*)d_in[3];
  const float* proj_w    = (const float*)d_in[4];
  const float* proj_b    = (const float*)d_in[5];
  const float* gat_wl    = (const float*)d_in[6];
  const float* gat_bl    = (const float*)d_in[7];
  const float* gat_wr    = (const float*)d_in[8];
  const float* gat_br    = (const float*)d_in[9];
  const float* gat_we    = (const float*)d_in[10];
  const float* gat_att   = (const float*)d_in[11];
  const float* gat_bias  = (const float*)d_in[12];
  const float* ln_g      = (const float*)d_in[13];
  const float* ln_b      = (const float*)d_in[14];
  const float* n2t_w     = (const float*)d_in[15];
  const float* n2t_b     = (const float*)d_in[16];
  const float* tf_wqkv   = (const float*)d_in[17];
  const float* tf_bqkv   = (const float*)d_in[18];
  const float* tf_wo     = (const float*)d_in[19];
  const float* tf_bo     = (const float*)d_in[20];
  const float* tf_w1     = (const float*)d_in[21];
  const float* tf_b1     = (const float*)d_in[22];
  const float* tf_w2     = (const float*)d_in[23];
  const float* tf_b2     = (const float*)d_in[24];
  const float* tf_ln1g   = (const float*)d_in[25];
  const float* tf_ln1b   = (const float*)d_in[26];
  const float* tf_ln2g   = (const float*)d_in[27];
  const float* tf_ln2b   = (const float*)d_in[28];
  float* out = (float*)d_out;

  // ---- workspace carve-up (floats) ----
  float* wsf = (float*)d_ws;
  size_t off = 0;
  float*    h       = wsf + off; off += (size_t)cT * cN * cG;
  float*    xl      = wsf + off; off += (size_t)cT * cN * cGH * cG;
  float*    xr      = wsf + off; off += (size_t)cT * cN * cGH * cG;   // aliased as acc
  float*    logit   = wsf + off; off += (size_t)cT * cE * cGH;
  unsigned* lmax    = (unsigned*)(wsf + off); off += (size_t)cT * cN * cGH;
  float*    denom   = wsf + off; off += (size_t)cT * cN * cGH;
  float*    meansum = wsf + off; off += (size_t)cT * cB * cG;
  unsigned* maxenc  = (unsigned*)(wsf + off); off += (size_t)cT * cB * cG;
  int*      counts  = (int*)(wsf + off); off += cB;
  float*    pooled  = wsf + off; off += (size_t)cT * cB * 2 * cG;
  float*    tokens  = wsf + off; off += (size_t)cT * cB * cTOK;
  float*    app     = wsf + off; off += (size_t)cB * cT * cTOK;
  float*    hbuf    = wsf + off; off += (size_t)cB * cT * cTOK;
  float*    qkv     = wsf + off; off += (size_t)cB * cT * 3 * cTOK;
  float*    attnout = wsf + off; off += (size_t)cB * cT * cTOK;
  float*    tmp512  = wsf + off; off += (size_t)cB * cT * cTOK;
  float*    ff      = wsf + off; off += (size_t)cB * cT * 4 * cTOK;
  float*    acc     = xr;   // xr is dead after edge_logit; reuse as aggregation buffer

  const int TN = cT * cN;
  const int TEH = cT * cE * cGH;

  // ---- stage 1: projection h = x @ proj_w + proj_b ----
  gemm_wmma_kernel<<<gemm_grid(TN, cG), 256, 0, stream>>>(x, proj_w, proj_b, h, TN, cIN, cG, 0);

  // ---- stage 2: GAT layers (batched over T) ----
  for (int l = 0; l < cLG; ++l) {
    const float* wl = gat_wl + (size_t)l * cG * cGH * cG;
    const float* bl = gat_bl + (size_t)l * cGH * cG;
    const float* wr = gat_wr + (size_t)l * cG * cGH * cG;
    const float* br = gat_br + (size_t)l * cGH * cG;
    const float* we = gat_we + (size_t)l * 2 * cGH * cG;
    const float* at = gat_att + (size_t)l * cGH * cG;
    const float* gb = gat_bias + (size_t)l * cG;
    const float* lg = ln_g + (size_t)l * cG;
    const float* lb = ln_b + (size_t)l * cG;

    gemm_wmma_kernel<<<gemm_grid(TN, cGH * cG), 256, 0, stream>>>(h, wl, bl, xl, TN, cG, cGH * cG, 0);
    gemm_wmma_kernel<<<gemm_grid(TN, cGH * cG), 256, 0, stream>>>(h, wr, br, xr, TN, cG, cGH * cG, 0);

    int nNH = TN * cGH;
    fill_u32_kernel<<<(nNH + 255) / 256, 256, 0, stream>>>(lmax, ORD_NEG_INF, nNH);
    fill_f32_kernel<<<(nNH + 255) / 256, 256, 0, stream>>>(denom, 0.f, nNH);

    edge_logit_kernel<<<(TEH + 255) / 256, 256, 0, stream>>>(edge_idx, edge_attr, we, at,
                                                             xl, xr, logit, lmax);
    int nAcc = TN * cGH * cG;
    fill_f32_kernel<<<(nAcc + 255) / 256, 256, 0, stream>>>(acc, 0.f, nAcc);

    edge_p_kernel<<<(TEH + 255) / 256, 256, 0, stream>>>(edge_idx, logit, lmax, denom);
    edge_aggr_kernel<<<cT * cE, cG, 0, stream>>>(edge_idx, logit, denom, xl, acc);
    node_update_kernel<<<TN, cG, 0, stream>>>(acc, gb, lg, lb, h);
  }

  // ---- stage 3: pooling + token projection ----
  fill_f32_kernel<<<(cT * cB * cG + 255) / 256, 256, 0, stream>>>(meansum, 0.f, cT * cB * cG);
  fill_u32_kernel<<<(cT * cB * cG + 255) / 256, 256, 0, stream>>>(maxenc, ORD_NEG_INF, cT * cB * cG);
  fill_u32_kernel<<<1, 256, 0, stream>>>((unsigned*)counts, 0u, cB);
  count_kernel<<<(cN + 255) / 256, 256, 0, stream>>>(batch, counts);
  pool_kernel<<<TN, cG, 0, stream>>>(h, batch, meansum, maxenc);
  pool_fin_kernel<<<(cT * cB * 2 * cG + 255) / 256, 256, 0, stream>>>(meansum, maxenc, counts, pooled);
  gemm_wmma_kernel<<<gemm_grid(cT * cB, cTOK), 256, 0, stream>>>(pooled, n2t_w, n2t_b, tokens,
                                                                cT * cB, 2 * cG, cTOK, 0);
  build_app_kernel<<<(cB * cT * cTOK + 255) / 256, 256, 0, stream>>>(tokens, app, hbuf);

  // ---- stage 4: transformer (rows = b*T+t, M=48) ----
  const int R = cB * cT;
  for (int l = 0; l < cLT; ++l) {
    const float* wq = tf_wqkv + (size_t)l * cTOK * 3 * cTOK;
    const float* bq = tf_bqkv + (size_t)l * 3 * cTOK;
    const float* wo = tf_wo + (size_t)l * cTOK * cTOK;
    const float* bo = tf_bo + (size_t)l * cTOK;
    const float* w1 = tf_w1 + (size_t)l * cTOK * 4 * cTOK;
    const float* b1 = tf_b1 + (size_t)l * 4 * cTOK;
    const float* w2 = tf_w2 + (size_t)l * 4 * cTOK * cTOK;
    const float* b2 = tf_b2 + (size_t)l * cTOK;

    gemm_wmma_kernel<<<gemm_grid(R, 3 * cTOK), 256, 0, stream>>>(hbuf, wq, bq, qkv, R, cTOK, 3 * cTOK, 0);
    attn_kernel<<<(cB * cNH * cT + 127) / 128, 128, 0, stream>>>(qkv, attnout);
    gemm_wmma_kernel<<<gemm_grid(R, cTOK), 256, 0, stream>>>(attnout, wo, bo, tmp512, R, cTOK, cTOK, 0);
    add_ln512_kernel<<<R, cTOK, 0, stream>>>(hbuf, tmp512,
                                             tf_ln1g + (size_t)l * cTOK, tf_ln1b + (size_t)l * cTOK);
    gemm_wmma_kernel<<<gemm_grid(R, 4 * cTOK), 256, 0, stream>>>(hbuf, w1, b1, ff, R, cTOK, 4 * cTOK, 1);
    gemm_wmma_kernel<<<gemm_grid(R, cTOK), 256, 0, stream>>>(ff, w2, b2, tmp512, R, 4 * cTOK, cTOK, 0);
    add_ln512_kernel<<<R, cTOK, 0, stream>>>(hbuf, tmp512,
                                             tf_ln2g + (size_t)l * cTOK, tf_ln2b + (size_t)l * cTOK);
  }

  // ---- outputs ----
  int totalOut = cB * cT * cTOK + cB * (cT - 1) * 256;
  out_kernel<<<(totalOut + 255) / 256, 256, 0, stream>>>(hbuf, app, out);
}
